// HiddenLayer_17695265259691
// MI455X (gfx1250) — compile-verified, matrix operand order checked by reading, not run on hardware
//
#include <hip/hip_runtime.h>
#include <hip/hip_bf16.h>

// ---------------- types for WMMA ----------------
typedef __bf16 v16bf __attribute__((ext_vector_type(16)));
typedef __bf16 v8bf  __attribute__((ext_vector_type(8)));
typedef float  v8f   __attribute__((ext_vector_type(8)));

union V16 { v16bf v; v8bf h[2]; };

// ---------------- problem dims (fixed by setup_inputs) ----------------
constexpr int NT  = 2016;   // Ntot rows of X_mean etc.
constexpr int Q   = 8;
constexpr int LT  = 16;
constexpr int M   = 128;    // inducing points
constexpr int NN  = 2000;   // hankel rows
constexpr int DK  = 256;    // 2*Lt*Q kernel dims
constexpr int K2  = 512;    // concatenated GEMM K
constexpr int MP  = M * M;  // 16384 column pairs
constexpr int NCH = NN / 16;  // 125 row chunks
constexpr int KCH = K2 / 32;  // 16 k chunks of 32 (bf16 WMMA K)

// ---------------- workspace layout (bytes) ----------------
constexpr size_t alup(size_t x) { return (x + 255) & ~size_t(255); }
constexpr size_t OFF_CMAT = 0;                                    // bf16 NN x K2
constexpr size_t OFF_UMAT = OFF_CMAT + alup((size_t)NN * K2 * 2); // bf16 NN x K2
constexpr size_t OFF_DMAT = OFF_UMAT + alup((size_t)NN * K2 * 2); // bf16 MP x K2
constexpr size_t OFF_ZZ   = OFF_DMAT + alup((size_t)MP * K2 * 2); // bf16 M x K2
constexpr size_t OFF_RF1  = OFF_ZZ   + alup((size_t)M * K2 * 2);  // f32 NN
constexpr size_t OFF_RF2  = OFF_RF1  + alup((size_t)NN * 4);      // f32 NN
constexpr size_t OFF_END  = OFF_RF2  + alup((size_t)NN * 4);      // f32 MP  exp(-distZ)
constexpr size_t OFF_KUU  = OFF_END  + alup((size_t)MP * 4);      // f32 M*M
constexpr size_t OFF_L    = OFF_KUU  + alup((size_t)M * M * 4);   // f32 M*M
constexpr size_t OFF_PSI1 = OFF_L    + alup((size_t)M * M * 4);   // f32 NN*M
constexpr size_t OFF_PSI2 = OFF_PSI1 + alup((size_t)NN * M * 4);  // f32 M*M
constexpr size_t OFF_T    = OFF_PSI2 + alup((size_t)M * M * 4);   // f32 NN*M  (L \ psi1^T)
constexpr size_t OFF_TMP  = OFF_T    + alup((size_t)NN * M * 4);  // f32 M*M
constexpr size_t OFF_AAT  = OFF_TMP  + alup((size_t)M * M * 4);   // f32 M*M
constexpr size_t OFF_LB   = OFF_AAT  + alup((size_t)M * M * 4);   // f32 M*M
constexpr size_t OFF_AX   = OFF_LB   + alup((size_t)M * M * 4);   // f32 M*Q
constexpr size_t OFF_SCAL = OFF_AX   + alup((size_t)M * Q * 4);   // f32 [trace, logdetB, sumc2]

__device__ __forceinline__ unsigned short f2bf(float f) {
  unsigned int u = __float_as_uint(f);
  u += 0x7FFFu + ((u >> 16) & 1u);      // round-to-nearest-even
  return (unsigned short)(u >> 16);
}

// ---------------- 1) per-row prep: hankel gather, bf16 staging, row factors ----------------
__global__ void prep_kernel(const float* __restrict__ Xm, const float* __restrict__ Xv,
                            const float* __restrict__ Mm, const float* __restrict__ Mv,
                            const float* __restrict__ ls,
                            unsigned short* __restrict__ Cmat, unsigned short* __restrict__ Umat,
                            float* __restrict__ rowfac1, float* __restrict__ rowfac2) {
  int n = blockIdx.x, t = threadIdx.x;   // n < NN, t < DK
  float xm, xv;
  if (t < 128) { int l = t >> 3, q = t & 7; xm = Xm[(n + l) * Q + q]; xv = Xv[(n + l) * Q + q]; }
  else { int tt = t - 128; int l = tt >> 3, q = tt & 7; xm = Mm[(1 + n + l) * Q + q]; xv = Mv[(1 + n + l) * Q + q]; }
  float lv = ls[t], l2 = lv * lv;
  float d1 = xv + l2, inv1 = 1.f / d1;
  float r1 = xm * xm * inv1;
  float g1 = 0.5f * (__logf(l2) - __logf(d1));
  float d2 = 2.f * xv + l2, w = 1.f / d2;
  float s2 = xm * xm * w;
  float g2 = 0.5f * (__logf(l2) - __logf(d2));
  size_t base = (size_t)n * K2;
  Umat[base + t]      = f2bf(xm * inv1);     // psi1 GEMM row half 1
  Umat[base + DK + t] = f2bf(inv1);          // psi1 GEMM row half 2
  Cmat[base + t]      = f2bf(-2.f * xm * w); // psi2 GEMM row half 1
  Cmat[base + DK + t] = f2bf(w);             // psi2 GEMM row half 2
  __shared__ float sA[256], sB[256], sC[256], sD[256];
  sA[t] = r1; sB[t] = g1; sC[t] = s2; sD[t] = g2;
  __syncthreads();
  for (int s = 128; s > 0; s >>= 1) {
    if (t < s) { sA[t] += sA[t + s]; sB[t] += sB[t + s]; sC[t] += sC[t + s]; sD[t] += sD[t + s]; }
    __syncthreads();
  }
  if (t == 0) { rowfac1[n] = sB[0] - 0.5f * sA[0]; rowfac2[n] = sD[0] - sC[0]; }
}

// ---------------- 2) Z-side columns for psi1 GEMM ----------------
__global__ void zz_kernel(const float* __restrict__ Z, unsigned short* __restrict__ ZZ) {
  int j = blockIdx.x, t = threadIdx.x;
  float z = Z[j * DK + t];
  ZZ[(size_t)j * K2 + t]      = f2bf(-2.f * z);
  ZZ[(size_t)j * K2 + DK + t] = f2bf(z * z);
}

// ---------------- 3) pair (m,p): Kuu, exp(-distZ), psi2 GEMM columns ----------------
__global__ void pair_kernel(const float* __restrict__ Z, const float* __restrict__ ls,
                            const float* __restrict__ kvp,
                            unsigned short* __restrict__ Dmat, float* __restrict__ expnd,
                            float* __restrict__ Kuu) {
  int mp = blockIdx.x; int m = mp >> 7, p = mp & 127; int t = threadIdx.x;
  float zm = Z[m * DK + t], zp = Z[p * DK + t];
  float lv = ls[t], l2 = lv * lv;
  float dz = zm - zp;
  float a = dz * dz / l2;           // distZ = sum(a)/4 ; Kuu exponent = -0.5*sum(a)
  float zb = 0.5f * (zm + zp);
  size_t base = (size_t)mp * K2;
  Dmat[base + t]      = f2bf(zb);
  Dmat[base + DK + t] = f2bf(zb * zb);
  __shared__ float s[256];
  s[t] = a; __syncthreads();
  for (int r = 128; r > 0; r >>= 1) { if (t < r) s[t] += s[t + r]; __syncthreads(); }
  if (t == 0) {
    float sA = s[0];
    expnd[mp] = __expf(-0.25f * sA);
    Kuu[mp]   = (*kvp) * __expf(-0.5f * sA) + ((m == p) ? 1e-6f : 0.f);
  }
}

// ---------------- 4) psi1 via WMMA bf16: 2000 x 128, K=512 ----------------
__global__ __launch_bounds__(32, 1)
void psi1_kernel(const unsigned short* __restrict__ Umat,
                 const unsigned short* __restrict__ ZZ,
                 const float* __restrict__ rowfac1, const float* __restrict__ kvp,
                 float* __restrict__ psi1) {
  int lane = threadIdx.x & 31;
  int j0 = blockIdx.x * 16;           // 8 strips of 16 columns
  int l16 = lane & 15, h = lane >> 4;
  float kv = *kvp;
  const v8bf* Brow = (const v8bf*)(ZZ + (size_t)(j0 + l16) * K2);
  V16 bfrag[KCH];
#pragma unroll
  for (int kc = 0; kc < KCH; ++kc) { bfrag[kc].h[0] = Brow[kc * 4 + h]; bfrag[kc].h[1] = Brow[kc * 4 + 2 + h]; }
  for (int nc = 0; nc < NCH; ++nc) {
    const v8bf* Arow = (const v8bf*)(Umat + (size_t)(nc * 16 + l16) * K2);
    if (nc + 1 < NCH)
      __builtin_prefetch((const char*)(Umat + (size_t)((nc + 1) * 16 + l16) * K2), 0, 3);
    V16 cur, nxt;
    cur.h[0] = Arow[h]; cur.h[1] = Arow[2 + h];
    v8f acc = {};
#pragma unroll
    for (int kc = 0; kc < KCH; ++kc) {
      if (kc + 1 < KCH) { nxt.h[0] = Arow[(kc + 1) * 4 + h]; nxt.h[1] = Arow[(kc + 1) * 4 + 2 + h]; }
      acc = __builtin_amdgcn_wmma_f32_16x16x32_bf16(false, cur.v, false, bfrag[kc].v,
                                                    (short)0, acc, false, false);
      cur = nxt;
    }
#pragma unroll
    for (int r = 0; r < 8; ++r) {
      int n = nc * 16 + r + 8 * h;   // C/D layout: VGPR r -> M = r + 8*(lane/16)
      psi1[(size_t)n * M + j0 + l16] = kv * __expf(rowfac1[n] - 0.5f * acc[r]);
    }
  }
}

// ---------------- 5) psi2 via WMMA bf16: GEMM 2000 x 16384 (K=512) + exp + n-reduction ------
// Two n-chunks per iteration -> two independent accumulator chains so WMMAs of one chunk
// overlap the in-flight A loads of the other (partial s_wait_loadcnt instead of drain-to-0).
__global__ __launch_bounds__(128, 1)
void psi2_kernel(const unsigned short* __restrict__ Cmat,
                 const unsigned short* __restrict__ Dmat,
                 const float* __restrict__ rowfac2, const float* __restrict__ expnd,
                 const float* __restrict__ kvp, float* __restrict__ psi2) {
  int wave = (blockIdx.x * blockDim.x + threadIdx.x) >> 5;  // 1024 waves, 16 cols each
  int lane = threadIdx.x & 31;
  int mp0 = wave * 16;
  int l16 = lane & 15, h = lane >> 4;
  const v8bf* Brow = (const v8bf*)(Dmat + (size_t)(mp0 + l16) * K2);
  V16 bfrag[KCH];
#pragma unroll
  for (int kc = 0; kc < KCH; ++kc) { bfrag[kc].h[0] = Brow[kc * 4 + h]; bfrag[kc].h[1] = Brow[kc * 4 + 2 + h]; }
  float csum = 0.f;
  const size_t rowstep = (size_t)K2;              // halves per row
  // ---- 62 pairs of chunks ----
  for (int ncp = 0; ncp < NCH / 2; ++ncp) {
    int nc0 = ncp * 2, nc1 = nc0 + 1;
    const v8bf* A0 = (const v8bf*)(Cmat + ((size_t)(nc0 * 16 + l16)) * rowstep);
    const v8bf* A1 = (const v8bf*)(Cmat + ((size_t)(nc1 * 16 + l16)) * rowstep);
    __builtin_prefetch((const char*)(Cmat + ((size_t)((nc0 + 2) * 16 + l16)) * rowstep), 0, 3);
    __builtin_prefetch((const char*)(Cmat + ((size_t)((nc1 + 2) * 16 + l16)) * rowstep), 0, 3);
    V16 c0, c1, x0, x1;
    c0.h[0] = A0[h]; c0.h[1] = A0[2 + h];
    c1.h[0] = A1[h]; c1.h[1] = A1[2 + h];
    v8f acc0 = {}, acc1 = {};
#pragma unroll
    for (int kc = 0; kc < KCH; ++kc) {
      if (kc + 1 < KCH) {
        x0.h[0] = A0[(kc + 1) * 4 + h]; x0.h[1] = A0[(kc + 1) * 4 + 2 + h];
        x1.h[0] = A1[(kc + 1) * 4 + h]; x1.h[1] = A1[(kc + 1) * 4 + 2 + h];
      }
      acc0 = __builtin_amdgcn_wmma_f32_16x16x32_bf16(false, c0.v, false, bfrag[kc].v,
                                                     (short)0, acc0, false, false);
      acc1 = __builtin_amdgcn_wmma_f32_16x16x32_bf16(false, c1.v, false, bfrag[kc].v,
                                                     (short)0, acc1, false, false);
      c0 = x0; c1 = x1;
    }
    // epilogue: exp + accumulate over the 8 rows each lane holds (per chunk)
    const float4* rf0 = (const float4*)(rowfac2 + nc0 * 16 + 8 * h);
    const float4* rf1 = (const float4*)(rowfac2 + nc1 * 16 + 8 * h);
    float4 ra = rf0[0], rb = rf0[1], rc = rf1[0], rd = rf1[1];
    csum += __expf(ra.x - acc0[0]) + __expf(ra.y - acc0[1]) +
            __expf(ra.z - acc0[2]) + __expf(ra.w - acc0[3]) +
            __expf(rb.x - acc0[4]) + __expf(rb.y - acc0[5]) +
            __expf(rb.z - acc0[6]) + __expf(rb.w - acc0[7]);
    csum += __expf(rc.x - acc1[0]) + __expf(rc.y - acc1[1]) +
            __expf(rc.z - acc1[2]) + __expf(rc.w - acc1[3]) +
            __expf(rd.x - acc1[4]) + __expf(rd.y - acc1[5]) +
            __expf(rd.z - acc1[6]) + __expf(rd.w - acc1[7]);
  }
  // ---- tail chunk (nc = 124) ----
  {
    int nc = NCH - 1;
    const v8bf* Arow = (const v8bf*)(Cmat + ((size_t)(nc * 16 + l16)) * rowstep);
    V16 cur, nxt;
    cur.h[0] = Arow[h]; cur.h[1] = Arow[2 + h];
    v8f acc = {};
#pragma unroll
    for (int kc = 0; kc < KCH; ++kc) {
      if (kc + 1 < KCH) { nxt.h[0] = Arow[(kc + 1) * 4 + h]; nxt.h[1] = Arow[(kc + 1) * 4 + 2 + h]; }
      acc = __builtin_amdgcn_wmma_f32_16x16x32_bf16(false, cur.v, false, bfrag[kc].v,
                                                    (short)0, acc, false, false);
      cur = nxt;
    }
    const float4* rf = (const float4*)(rowfac2 + nc * 16 + 8 * h);
    float4 ra = rf[0], rb = rf[1];
    csum += __expf(ra.x - acc[0]) + __expf(ra.y - acc[1]) +
            __expf(ra.z - acc[2]) + __expf(ra.w - acc[3]) +
            __expf(rb.x - acc[4]) + __expf(rb.y - acc[5]) +
            __expf(rb.z - acc[6]) + __expf(rb.w - acc[7]);
  }
  csum += __shfl_xor(csum, 16, 32);   // lanes L and L+16 hold same column N
  if (h == 0) {
    float kv = *kvp;
    psi2[mp0 + l16] = kv * kv * expnd[mp0 + l16] * csum;
  }
}

// ---------------- 6) Cholesky of Kuu (single block, in global) ----------------
__global__ void chol_kernel(const float* __restrict__ Ain, float* __restrict__ Lm) {
  int tid = threadIdx.x;  // 128 threads, one row each
  for (int j = 0; j < M; ++j) Lm[tid * M + j] = (j <= tid) ? Ain[tid * M + j] : 0.f;
  __syncthreads();
  for (int k = 0; k < M; ++k) {
    if (tid == k) Lm[k * M + k] = sqrtf(Lm[k * M + k]);
    __syncthreads();
    if (tid > k) Lm[tid * M + k] /= Lm[k * M + k];
    __syncthreads();
    if (tid > k) { float lik = Lm[tid * M + k]; for (int j = k + 1; j <= tid; ++j) Lm[tid * M + j] -= lik * Lm[j * M + k]; }
    __syncthreads();
  }
}

// ---------------- 7) T = L \ psi1^T  (one thread per column n) ----------------
__global__ void trsm1_kernel(const float* __restrict__ Lm, const float* __restrict__ psi1,
                             float* __restrict__ T) {
  int n = blockIdx.x * blockDim.x + threadIdx.x;
  if (n >= NN) return;
  float x[M];
  for (int i = 0; i < M; ++i) {
    float acc = psi1[(size_t)n * M + i];
    for (int j = 0; j < i; ++j) acc -= Lm[i * M + j] * x[j];
    x[i] = acc / Lm[i * M + i];
    T[(size_t)n * M + i] = x[i];
  }
}

// ---------------- 8a/8b) tmp = L \ psi2 ; AAT = (L \ tmp^T)/sigma2 ----------------
__global__ void trsm2a_kernel(const float* __restrict__ Lm, const float* __restrict__ psi2,
                              float* __restrict__ tmpM) {
  int c = threadIdx.x;
  float x[M];
  for (int i = 0; i < M; ++i) {
    float acc = psi2[i * M + c];
    for (int j = 0; j < i; ++j) acc -= Lm[i * M + j] * x[j];
    x[i] = acc / Lm[i * M + i];
    tmpM[i * M + c] = x[i];
  }
}
__global__ void trsm2b_kernel(const float* __restrict__ Lm, const float* __restrict__ tmpM,
                              const float* __restrict__ livp, float* __restrict__ AAT) {
  int c = threadIdx.x;
  float s2 = *livp;
  float x[M];
  for (int i = 0; i < M; ++i) {
    float acc = tmpM[c * M + i];
    for (int j = 0; j < i; ++j) acc -= Lm[i * M + j] * x[j];
    x[i] = acc / Lm[i * M + i];
    AAT[i * M + c] = x[i] / s2;
  }
}

// ---------------- 9) B = AAT + I; LB = chol(B); trace(AAT), log|B| ----------------
__global__ void cholb_kernel(const float* __restrict__ AAT, float* __restrict__ LB,
                             float* __restrict__ scal) {
  int tid = threadIdx.x;  // 128
  __shared__ float red[128];
  for (int j = 0; j < M; ++j)
    LB[tid * M + j] = (j < tid) ? AAT[tid * M + j] : ((j == tid) ? AAT[tid * M + j] + 1.f : 0.f);
  red[tid] = AAT[tid * M + tid];
  __syncthreads();
  for (int s = 64; s > 0; s >>= 1) { if (tid < s) red[tid] += red[tid + s]; __syncthreads(); }
  if (tid == 0) scal[0] = red[0];                    // trace(AAT)
  __syncthreads();
  for (int k = 0; k < M; ++k) {
    if (tid == k) LB[k * M + k] = sqrtf(LB[k * M + k]);
    __syncthreads();
    if (tid > k) LB[tid * M + k] /= LB[k * M + k];
    __syncthreads();
    if (tid > k) { float lik = LB[tid * M + k]; for (int j = k + 1; j <= tid; ++j) LB[tid * M + j] -= lik * LB[j * M + k]; }
    __syncthreads();
  }
  red[tid] = __logf(LB[tid * M + tid]);
  __syncthreads();
  for (int s = 64; s > 0; s >>= 1) { if (tid < s) red[tid] += red[tid + s]; __syncthreads(); }
  if (tid == 0) scal[1] = 2.f * red[0];              // log_det_B
}

// ---------------- 10) AX = T^T @ X_mo  (128 x 8) ----------------
__global__ void axo_kernel(const float* __restrict__ T, const float* __restrict__ Xmean,
                           float* __restrict__ AX) {
  int id = blockIdx.x * blockDim.x + threadIdx.x;
  if (id >= M * Q) return;
  int i = id >> 3, d = id & 7;
  float s = 0.f;
  for (int n = 0; n < NN; ++n) s += T[(size_t)n * M + i] * Xmean[(LT + n) * Q + d];
  AX[i * Q + d] = s;
}

// ---------------- 11) c = (LB \ AX)/sigma2 ; sum(c^2) ----------------
__global__ void csolve_kernel(const float* __restrict__ LB, const float* __restrict__ AX,
                              const float* __restrict__ livp, float* __restrict__ scal) {
  int d = threadIdx.x;  // 32 threads, 8 active
  __shared__ float red[32];
  float part = 0.f;
  if (d < Q) {
    float s2 = *livp;
    float x[M];
    for (int i = 0; i < M; ++i) {
      float acc = AX[i * Q + d];
      for (int j = 0; j < i; ++j) acc -= LB[i * M + j] * x[j];
      x[i] = acc / LB[i * M + i];
      float c = x[i] / s2;
      part += c * c;
    }
  }
  red[d] = part; __syncthreads();
  for (int s = 16; s > 0; s >>= 1) { if (d < s) red[d] += red[d + s]; __syncthreads(); }
  if (d == 0) scal[2] = red[0];
}

// ---------------- 12) scalar assembly ----------------
__global__ void finalize_kernel(const float* __restrict__ Xmean, const float* __restrict__ Xvar,
                                const float* __restrict__ kvp, const float* __restrict__ livp,
                                const float* __restrict__ scal, float* __restrict__ out) {
  int t = threadIdx.x;
  float slog = 0.f, svo = 0.f, smo2 = 0.f, sb = 0.f;
  for (int idx = t; idx < NN * Q; idx += 256) {
    float xv = Xvar[LT * Q + idx], xm = Xmean[LT * Q + idx];
    slog += __logf(xv); svo += xv; smo2 += xm * xm;
  }
  if (t < LT * Q) { float xm = Xmean[t], xv = Xvar[t]; sb = xm * xm + xv; }
  __shared__ float rA[256], rB[256], rC[256], rD[256];
  rA[t] = slog; rB[t] = svo; rC[t] = smo2; rD[t] = sb;
  __syncthreads();
  for (int s = 128; s > 0; s >>= 1) {
    if (t < s) { rA[t] += rA[t + s]; rB[t] += rB[t + s]; rC[t] += rC[t + s]; rD[t] += rD[t + s]; }
    __syncthreads();
  }
  if (t == 0) {
    const float LOG2PI = 1.83787706640934548f;
    float kv = *kvp, s2 = *livp;
    float tr = scal[0], ldB = scal[1], sc2 = scal[2];
    float psi0 = (float)NN * kv;
    float ent  = 0.5f * rA[0] + (float)NN * Q * 0.5f * LOG2PI;
    float ent2 = -(float)(LT * Q) * LOG2PI - 0.5f * rD[0];
    float bound = -0.5f * (float)NN * Q * (LOG2PI + __logf(s2));
    bound += -0.5f / s2 * (rB[0] + rC[0]);
    bound += -0.5f * Q * (psi0 / s2 - tr);
    bound += -0.5f * Q * ldB;
    bound += 0.5f * sc2;
    bound += ent + ent2;
    out[0] = bound;
  }
}

// ---------------- launch ----------------
extern "C" void kernel_launch(void* const* d_in, const int* in_sizes, int n_in,
                              void* d_out, int out_size, void* d_ws, size_t ws_size,
                              hipStream_t stream) {
  const float* Z    = (const float*)d_in[0];
  const float* Xm   = (const float*)d_in[1];
  const float* Xv   = (const float*)d_in[2];
  const float* kvp  = (const float*)d_in[3];
  const float* ls   = (const float*)d_in[4];
  const float* livp = (const float*)d_in[5];
  const float* Mm   = (const float*)d_in[6];
  const float* Mv   = (const float*)d_in[7];

  char* ws = (char*)d_ws;
  unsigned short* Cmat = (unsigned short*)(ws + OFF_CMAT);
  unsigned short* Umat = (unsigned short*)(ws + OFF_UMAT);
  unsigned short* Dmat = (unsigned short*)(ws + OFF_DMAT);
  unsigned short* ZZ   = (unsigned short*)(ws + OFF_ZZ);
  float* rowfac1 = (float*)(ws + OFF_RF1);
  float* rowfac2 = (float*)(ws + OFF_RF2);
  float* expnd   = (float*)(ws + OFF_END);
  float* Kuu     = (float*)(ws + OFF_KUU);
  float* Lm      = (float*)(ws + OFF_L);
  float* psi1    = (float*)(ws + OFF_PSI1);
  float* psi2    = (float*)(ws + OFF_PSI2);
  float* T       = (float*)(ws + OFF_T);
  float* tmpM    = (float*)(ws + OFF_TMP);
  float* AAT     = (float*)(ws + OFF_AAT);
  float* LB      = (float*)(ws + OFF_LB);
  float* AX      = (float*)(ws + OFF_AX);
  float* scal    = (float*)(ws + OFF_SCAL);

  prep_kernel<<<NN, 256, 0, stream>>>(Xm, Xv, Mm, Mv, ls, Cmat, Umat, rowfac1, rowfac2);
  zz_kernel<<<M, DK, 0, stream>>>(Z, ZZ);
  pair_kernel<<<MP, DK, 0, stream>>>(Z, ls, kvp, Dmat, expnd, Kuu);
  psi1_kernel<<<M / 16, 32, 0, stream>>>(Umat, ZZ, rowfac1, kvp, psi1);
  psi2_kernel<<<MP / 16 / 4, 128, 0, stream>>>(Cmat, Dmat, rowfac2, expnd, kvp, psi2);
  chol_kernel<<<1, M, 0, stream>>>(Kuu, Lm);
  trsm1_kernel<<<(NN + 255) / 256, 256, 0, stream>>>(Lm, psi1, T);
  trsm2a_kernel<<<1, M, 0, stream>>>(Lm, psi2, tmpM);
  trsm2b_kernel<<<1, M, 0, stream>>>(Lm, tmpM, livp, AAT);
  cholb_kernel<<<1, M, 0, stream>>>(AAT, LB, scal);
  axo_kernel<<<(M * Q + 255) / 256, 256, 0, stream>>>(T, Xm, AX);
  csolve_kernel<<<1, 32, 0, stream>>>(LB, AX, livp, scal);
  finalize_kernel<<<1, 256, 0, stream>>>(Xm, Xv, kvp, livp, scal, (float*)d_out);
}
